// BitLinear_2757369004076
// MI455X (gfx1250) — 8-bit matrix operand order repaired and checked by reading, compiled, not run on hardware
//
#include <hip/hip_runtime.h>

typedef __attribute__((ext_vector_type(8))) int v8i;
typedef __attribute__((ext_vector_type(4))) int v4i;

#define Q_EPS    1e-5f
#define EPS_NORM 1e-5f

__global__ __launch_bounds__(256) void wabs_partial_kernel(
    const float* __restrict__ w, float* __restrict__ partials,
    unsigned long long n4)
{
    __shared__ float red[256];
    float s = 0.0f;
    const unsigned long long stride =
        (unsigned long long)gridDim.x * blockDim.x;
    for (unsigned long long i =
             (unsigned long long)blockIdx.x * blockDim.x + threadIdx.x;
         i < n4; i += stride) {
        float4 v = ((const float4*)w)[i];
        s += fabsf(v.x) + fabsf(v.y) + fabsf(v.z) + fabsf(v.w);
    }
    red[threadIdx.x] = s;
    __syncthreads();
    for (int k = 128; k > 0; k >>= 1) {
        if (threadIdx.x < k) red[threadIdx.x] += red[threadIdx.x + k];
        __syncthreads();
    }
    if (threadIdx.x == 0) partials[blockIdx.x] = red[0];
}

__global__ __launch_bounds__(256) void wscale_kernel(
    const float* __restrict__ partials, float* __restrict__ scales,
    float invCount)
{
    __shared__ float red[256];
    red[threadIdx.x] = partials[threadIdx.x];
    __syncthreads();
    for (int k = 128; k > 0; k >>= 1) {
        if (threadIdx.x < k) red[threadIdx.x] += red[threadIdx.x + k];
        __syncthreads();
    }
    if (threadIdx.x == 0) {
        float mean = red[0] * invCount;
        scales[0] = 1.0f / fmaxf(mean, Q_EPS);
    }
}

__global__ __launch_bounds__(256) void wquant_kernel(
    const float* __restrict__ w, const float* __restrict__ scales,
    signed char* __restrict__ wq, unsigned long long n4)
{
    const float sw = scales[0];
    const unsigned long long stride =
        (unsigned long long)gridDim.x * blockDim.x;
    for (unsigned long long i =
             (unsigned long long)blockIdx.x * blockDim.x + threadIdx.x;
         i < n4; i += stride) {
        float4 v = ((const float4*)w)[i];
        char4 q;
        q.x = (signed char)(int)fminf(fmaxf(rintf(v.x * sw), -1.0f), 1.0f);
        q.y = (signed char)(int)fminf(fmaxf(rintf(v.y * sw), -1.0f), 1.0f);
        q.z = (signed char)(int)fminf(fmaxf(rintf(v.z * sw), -1.0f), 1.0f);
        q.w = (signed char)(int)fminf(fmaxf(rintf(v.w * sw), -1.0f), 1.0f);
        ((char4*)wq)[i] = q;
    }
}

__global__ __launch_bounds__(256) void act_quant_kernel(
    const float* __restrict__ x, const float* __restrict__ nw,
    const float* __restrict__ scales, signed char* __restrict__ hq,
    float* __restrict__ dq, int K)
{
    __shared__ float sp[4096];
    __shared__ float red[256];

    const int tid = threadIdx.x;
    const int row = blockIdx.x;
    const float* __restrict__ xr = x + (size_t)row * K;

    float ss = 0.0f, amax = 0.0f;
    const int k4 = K >> 2;
    for (int i = tid; i < k4; i += blockDim.x) {
        float4 xv = ((const float4*)xr)[i];
        float4 wv = ((const float4*)nw)[i];
        float p0 = xv.x * wv.x, p1 = xv.y * wv.y;
        float p2 = xv.z * wv.z, p3 = xv.w * wv.w;
        ss += xv.x * xv.x + xv.y * xv.y + xv.z * xv.z + xv.w * xv.w;
        amax = fmaxf(amax, fmaxf(fmaxf(fabsf(p0), fabsf(p1)),
                                 fmaxf(fabsf(p2), fabsf(p3))));
        ((float4*)sp)[i] = make_float4(p0, p1, p2, p3);
    }

    red[tid] = ss;
    __syncthreads();
    for (int k = 128; k > 0; k >>= 1) {
        if (tid < k) red[tid] += red[tid + k];
        __syncthreads();
    }
    const float ssum = red[0];
    __syncthreads();

    red[tid] = amax;
    __syncthreads();
    for (int k = 128; k > 0; k >>= 1) {
        if (tid < k) red[tid] = fmaxf(red[tid], red[tid + k]);
        __syncthreads();
    }
    const float pmax = red[0];

    const float rs = rsqrtf(ssum / (float)K + EPS_NORM);
    const float sx = 127.0f / fmaxf(pmax * rs, Q_EPS);
    if (tid == 0) dq[row] = 1.0f / (sx * scales[0]);

    const float qs = rs * sx;
    signed char* __restrict__ hr = hq + (size_t)row * K;
    for (int i = tid; i < k4; i += blockDim.x) {
        float4 p = ((const float4*)sp)[i];
        char4 q;
        q.x = (signed char)min(127, max(-128, (int)rintf(p.x * qs)));
        q.y = (signed char)min(127, max(-128, (int)rintf(p.y * qs)));
        q.z = (signed char)min(127, max(-128, (int)rintf(p.z * qs)));
        q.w = (signed char)min(127, max(-128, (int)rintf(p.w * qs)));
        ((char4*)hr)[i] = q;
    }
}

#define BM 128
#define BN 256
#define BK 64

__global__ __launch_bounds__(512) void bitlinear_wmma_gemm(
    const signed char* __restrict__ hq, const signed char* __restrict__ wq,
    const float* __restrict__ dq, float* __restrict__ out,
    int M, int N, int K)
{
    __shared__ v4i As[2][BM * BK / 16];
    __shared__ v4i Bs[2][BN * BK / 16];

    const int tid  = threadIdx.x;
    const int wid  = tid >> 5;
    const int lane = tid & 31;
    const int half = lane >> 4;
    const int lrow = lane & 15;
    const int wm   = wid >> 3;
    const int wn   = wid & 7;

    const int mBlk = blockIdx.y * BM;
    const int nBlk = blockIdx.x * BN;

    const int rr = tid >> 2;
    const int cc = (tid & 3) * 16;
    const signed char* gA  = hq + (size_t)(mBlk + rr) * K + cc;
    const signed char* gB0 = wq + (size_t)(nBlk + rr) * K + cc;
    const signed char* gB1 = wq + (size_t)(nBlk + 128 + rr) * K + cc;

    v8i c[4][2] = {};
    union AU { v8i v; unsigned long long q[4]; };
    union BU { v8i v; v4i h[2]; };

    v4i ra  = *(const v4i*)(gA);
    v4i rb0 = *(const v4i*)(gB0);
    v4i rb1 = *(const v4i*)(gB1);
    As[0][tid] = ra;
    Bs[0][tid] = rb0;
    Bs[0][tid + 512] = rb1;
    __syncthreads();

    for (int kk = 0; kk < K; kk += BK) {
        const int  buf  = (kk >> 6) & 1;
        const bool more = (kk + BK) < K;

        if (more) {
            ra  = *(const v4i*)(gA  + kk + BK);
            rb0 = *(const v4i*)(gB0 + kk + BK);
            rb1 = *(const v4i*)(gB1 + kk + BK);
        }

        const char* as = (const char*)As[buf];
        const char* bs = (const char*)Bs[buf];

        AU a[4];
        #pragma unroll
        for (int i = 0; i < 4; ++i) {
            const char* p = as + (wm * 64 + i * 16 + lrow) * 64 + 16 * half;
            *(v4i*)&a[i].q[0] = *(const v4i*)(p);
            *(v4i*)&a[i].q[2] = *(const v4i*)(p + 32);
        }

        #pragma unroll
        for (int j = 0; j < 2; ++j) {
            BU b;
            const char* p = bs + (wn * 32 + j * 16 + lrow) * 64 + 16 * half;
            b.h[0] = *(const v4i*)(p);
            b.h[1] = *(const v4i*)(p + 32);
            #pragma unroll
            for (int i = 0; i < 4; ++i)
                c[i][j] = __builtin_amdgcn_wmma_i32_16x16x64_iu8(
                    true, a[i].v, true, b.v, c[i][j], false, false);
        }

        if (more) {
            As[buf ^ 1][tid] = ra;
            Bs[buf ^ 1][tid] = rb0;
            Bs[buf ^ 1][tid + 512] = rb1;
        }
        __syncthreads();
    }

    const int mW = mBlk + wm * 64;
    const int nW = nBlk + wn * 32 + lrow;
    #pragma unroll
    for (int i = 0; i < 4; ++i) {
        const int mrow = mW + i * 16 + 8 * half;
        #pragma unroll
        for (int v = 0; v < 8; ++v) {
            const float d = dq[mrow + v];
            float* __restrict__ orow = out + (size_t)(mrow + v) * N + nW;
            orow[0]  = (float)c[i][0][v] * d;
            orow[16] = (float)c[i][1][v] * d;
        }
    }
}

extern "C" void kernel_launch(void* const* d_in, const int* in_sizes, int n_in,
                              void* d_out, int out_size, void* d_ws, size_t ws_size,
                              hipStream_t stream) {
    const float* x  = (const float*)d_in[0];
    const float* nw = (const float*)d_in[1];
    const float* w  = (const float*)d_in[2];
    float* out = (float*)d_out;

    const int D = in_sizes[1];
    const int K = D;
    const int N = in_sizes[2] / D;
    const int M = in_sizes[0] / D;

    char* ws = (char*)d_ws;
    float*       partials = (float*)ws;
    float*       scales   = (float*)(ws + 1024);
    signed char* wq       = (signed char*)(ws + 4096);
    size_t off = 4096 + (size_t)N * K;
    off = (off + 4095) & ~(size_t)4095;
    float*       dq       = (float*)(ws + off);
    off += (size_t)M * sizeof(float);
    off = (off + 4095) & ~(size_t)4095;
    signed char* hq       = (signed char*)(ws + off);

    const unsigned long long wElems = (unsigned long long)N * K;

    wabs_partial_kernel<<<256, 256, 0, stream>>>(w, partials, wElems >> 2);
    wscale_kernel<<<1, 256, 0, stream>>>(partials, scales,
                                         1.0f / (float)wElems);
    wquant_kernel<<<256, 256, 0, stream>>>(w, scales, wq, wElems >> 2);
    act_quant_kernel<<<M, 256, 0, stream>>>(x, nw, scales, hq, dq, K);
    dim3 grid(N / BN, M / BM);
    bitlinear_wmma_gemm<<<grid, 512, 0, stream>>>(hq, wq, dq, out, M, N, K);
}
